// DetModel_37469294690803
// MI455X (gfx1250) — matrix operand order repaired and checked by reading, not compiled, not run on hardware
//
#include <hip/hip_runtime.h>
#include <hip/hip_bf16.h>

// ---------------------------------------------------------------------------
// DetNet MIMO detector on MI455X (gfx1250), wave32 + WMMA f16->f32 + TDM.
// B=8192, S=2*TX=64, FEAT=65 (padded to 96), HID=128, 3H=384, DIMZ=64.
// ---------------------------------------------------------------------------

#define BN_   8192
#define SQ_   64
#define FP_   96      // FEAT=65 padded to multiple of 32
#define HID_  128
#define G3_   384

typedef __attribute__((ext_vector_type(16))) _Float16 v16h;
typedef __attribute__((ext_vector_type(8)))  _Float16 v8h;
typedef __attribute__((ext_vector_type(8)))  float    v8f;
typedef __attribute__((ext_vector_type(4)))  unsigned int u32x4;
typedef __attribute__((ext_vector_type(8)))  int       i32x8;
typedef __attribute__((ext_vector_type(4)))  int       i32x4;

union AFrag { v16h v; v8h h[2]; };

// D = A*B + C  (16x16x32, f16 operands, f32 accumulate)
__device__ inline v8f wmma16(v16h a, v16h b, v8f c) {
  return __builtin_amdgcn_wmma_f32_16x16x32_f16(false, a, false, b, (short)0, c,
                                                false, false);
}

// A fragment (16x32 f16) from row-major [M x ld] storage (two 16B loads).
// Layout used here (contiguous k): lanes 0-15 hold row m, K={0..15}; lanes 16-31: K={16..31}.
__device__ inline v16h load_fragA_rm(const _Float16* base, int m0, int ld, int kt) {
  int lane = threadIdx.x & 31;
  int m    = m0 + (lane & 15);
  int kh   = (lane < 16) ? 0 : 16;
  const _Float16* p = base + (size_t)m * ld + kt * 32 + kh;
  AFrag f;
  f.h[0] = *(const v8h*)p;          // K = kh .. kh+7
  f.h[1] = *(const v8h*)(p + 8);    // K = kh+8 .. kh+15
  return f.v;
}

// B fragment from pre-swizzled pack: contiguous 16 f16 (32B) per lane.
// Convention: lane = (klocal>=16 ? 16 : 0) + (n&15); element i -> klocal = (lane<16?0:16)+i.
__device__ inline v16h load_fragB_pack(const _Float16* pack, int kt, int nt, int ntiles) {
  int lane = threadIdx.x & 31;
  return *(const v16h*)(pack + ((size_t)(kt * ntiles + nt) * 32 + lane) * 16);
}

// C/D layout: VGPR i, lanes 0-15 -> (M=m0+i, N=n0+lane); lanes 16-31 -> M=m0+8+i.
__device__ inline void store_acc_f16(_Float16* dst, int m0, int n0, int ld, v8f acc) {
  int lane = threadIdx.x & 31;
  int n    = n0 + (lane & 15);
  int mo   = (lane < 16) ? 0 : 8;
#pragma unroll
  for (int i = 0; i < 8; ++i) dst[(size_t)(m0 + mo + i) * ld + n] = (_Float16)acc[i];
}
__device__ inline void store_acc_f32(float* dst, int m0, int n0, int ld, v8f acc) {
  int lane = threadIdx.x & 31;
  int n    = n0 + (lane & 15);
  int mo   = (lane < 16) ? 0 : 8;
#pragma unroll
  for (int i = 0; i < 8; ++i) dst[(size_t)(m0 + mo + i) * ld + n] = acc[i];
}
// Store accumulator directly into B-fragment pack layout (ntiles = 4, 64x64 mat).
__device__ inline void store_acc_Bpack(_Float16* pack, int m0, int n0, v8f acc) {
  int lane = threadIdx.x & 31;
  int n    = n0 + (lane & 15);
  int mo   = (lane < 16) ? 0 : 8;
#pragma unroll
  for (int i = 0; i < 8; ++i) {
    int m  = m0 + mo + i;                    // k index of the pack
    int kt = m >> 5, ui = n >> 4;
    int pl = ((m >> 4) & 1) * 16 + (n & 15);
    pack[((size_t)(kt * 4 + ui) * 32 + pl) * 16 + (m & 15)] = (_Float16)acc[i];
  }
}

// ---------------------------------------------------------------------------
// K0: zero BN accumulators + pre-swizzle weight matrices into WMMA B-fragment
//     order (f32 -> f16).
// ---------------------------------------------------------------------------
#define NIH_ (2 * 3 * 24 * 512)
#define NHH_ (2 * 4 * 24 * 512)
#define NWX_ (8 * 4 * 512)

__global__ void k0_pack(const float* __restrict__ wih, const float* __restrict__ whh,
                        const float* __restrict__ Wx,
                        _Float16* __restrict__ wih_pack, _Float16* __restrict__ whh_pack,
                        _Float16* __restrict__ wx_pack, float* __restrict__ bn_acc) {
  int idx = blockIdx.x * 256 + threadIdx.x;
  if (idx < 128) bn_acc[idx] = 0.f;
  int e = idx - 128;
  if (e < 0) return;
  if (e < NIH_) {
    int i = e & 15, lane = (e >> 4) & 31;
    int rest = e >> 9;
    int nt = rest % 24; rest /= 24;
    int kt = rest % 3;  int d = rest / 3;
    int kloc = ((lane < 16) ? 0 : 16) + i;
    int k = kt * 32 + kloc;
    int g = nt * 16 + (lane & 15);
    float v = (k < 65) ? wih[((size_t)d * G3_ + g) * 65 + k] : 0.f;
    wih_pack[e] = (_Float16)v;
    return;
  }
  e -= NIH_;
  if (e < NHH_) {
    int i = e & 15, lane = (e >> 4) & 31;
    int rest = e >> 9;
    int nt = rest % 24; rest /= 24;
    int kt = rest % 4;  int d = rest / 4;
    int kloc = ((lane < 16) ? 0 : 16) + i;
    int k = kt * 32 + kloc;
    int g = nt * 16 + (lane & 15);
    whh_pack[e] = (_Float16)whh[((size_t)d * G3_ + g) * HID_ + k];
    return;
  }
  e -= NHH_;
  if (e < NWX_) {
    int i = e & 15, lane = (e >> 4) & 31;
    int rest = e >> 9;
    int nt = rest & 3, kt = rest >> 2;
    int kloc = ((lane < 16) ? 0 : 16) + i;
    int k = kt * 32 + kloc;
    int n = nt * 16 + (lane & 15);
    wx_pack[e] = (_Float16)Wx[(size_t)n * 256 + k];
  }
}

// ---------------------------------------------------------------------------
// K1: per-batch HtH (WMMA), Hty, BN partial sums. H tile (contiguous 16 KB)
//     is DMA'd into LDS with the Tensor Data Mover, then converted into the
//     two WMMA staging layouts (A row-major of H^T, B-fragment pack of H).
// ---------------------------------------------------------------------------
__global__ void __launch_bounds__(256) k1_prep(
    const float* __restrict__ H, const float* __restrict__ y,
    _Float16* __restrict__ HtH_h, float* __restrict__ Hty, float* __restrict__ bn_acc) {
  __shared__ __align__(16) float    Hf[64 * 64];    // raw H via TDM (16 KB)
  __shared__ __align__(16) _Float16 HhT[64 * 64];   // H^T row-major (A layout)
  __shared__ __align__(16) _Float16 Hpk[64 * 64];   // H in B-fragment pack
  __shared__ __align__(16) _Float16 Cl[64 * 64];    // HtH staging
  __shared__ float yl[64];
  __shared__ float htyl[64];
  int b = blockIdx.x, tid = threadIdx.x;
  const float* Hb = H + (size_t)b * 4096;
  if (tid < 64) yl[tid] = y[(size_t)b * 64 + tid];

  if (tid < 32) {   // one wave issues the TDM DMA: 1-D tile, 4096 x f32
    unsigned long long ga = (unsigned long long)(uintptr_t)Hb;
    unsigned lds = (unsigned)(uintptr_t)(&Hf[0]);
    u32x4 g0;
    g0[0] = 1u;                                              // count=1, no gather
    g0[1] = lds;                                             // lds_addr
    g0[2] = (unsigned)(ga & 0xffffffffu);                    // global_addr[31:0]
    g0[3] = (unsigned)((ga >> 32) & 0x1ffffffu) | (2u << 30);// addr[56:32] | type=2
    i32x8 g1;
    g1[0] = (int)(2u << 16);       // workgroup_mask=0, data_size=2 (4 bytes)
    g1[1] = (int)(4096u << 16);    // tensor_dim0 = 4096 (lo16 in [31:16])
    g1[2] = (int)(1u << 16);       // tensor_dim0 hi=0, tensor_dim1 = 1
    g1[3] = (int)(4096u << 16);    // tensor_dim1 hi=0, tile_dim0 = 4096
    g1[4] = 0;                     // tile_dim1 = tile_dim2 = 0 (unused)
    g1[5] = 4096;                  // tensor_dim0_stride lo32
    g1[6] = 0;
    g1[7] = 0;
    i32x4 gz4 = {0, 0, 0, 0};
    i32x8 gz8 = {0, 0, 0, 0, 0, 0, 0, 0};
    __builtin_amdgcn_tensor_load_to_lds(g0, g1, gz4, gz4, gz8, 0);
    __builtin_amdgcn_s_wait_tensorcnt(0);
  }
  __syncthreads();

  // Convert f32 H -> f16 staging layouts (vector-load friendly for WMMA).
  for (int i = tid; i < 4096; i += 256) {
    float v = Hf[i];
    int r = i >> 6, t = i & 63;
    HhT[t * 64 + r] = (_Float16)v;
    int kt = r >> 5, ui = t >> 4;
    int pl = ((r >> 4) & 1) * 16 + (t & 15);
    Hpk[((kt * 4 + ui) * 32 + pl) * 16 + (r & 15)] = (_Float16)v;
  }
  __syncthreads();

  if (tid < 64) {             // Hty[t] = sum_r H[r][t]*y[r]  (f32, from LDS)
    float s = 0.f;
    for (int r = 0; r < 64; ++r) s += Hf[r * 64 + tid] * yl[r];
    htyl[tid] = s;
    Hty[(size_t)b * 64 + tid] = s;
  }
  int wave = tid >> 5;
#pragma unroll
  for (int rep = 0; rep < 2; ++rep) {   // 16 output tiles / 8 waves
    int tile = wave * 2 + rep;
    int ti = tile >> 2, ui = tile & 3;
    v8f acc = {};
#pragma unroll
    for (int kt = 0; kt < 2; ++kt) {
      v16h a  = load_fragA_rm(HhT, ti * 16, 64, kt);   // A = H^T
      v16h bf = load_fragB_pack(Hpk, kt, ui, 4);       // B = H
      acc = wmma16(a, bf, acc);
    }
    store_acc_f16(Cl, ti * 16, ui * 16, 64, acc);
  }
  __syncthreads();
  _Float16* outb = HtH_h + (size_t)b * 4096;
  for (int i = tid; i < 4096; i += 256) outb[i] = Cl[i];
  if (tid < 64) {             // per-channel partial BN stats over 65 feats
    float s = htyl[tid], q = htyl[tid] * htyl[tid];
    for (int u = 0; u < 64; ++u) {
      float v = (float)Cl[tid * 64 + u];
      s += v; q += v * v;
    }
    atomicAdd(&bn_acc[tid], s);
    atomicAdd(&bn_acc[64 + tid], q);
  }
}

// K2: finalize BN -> per-channel scale/shift (folds gamma/beta).
__global__ void k2_bn(const float* __restrict__ bn_acc, const float* __restrict__ gamma,
                      const float* __restrict__ beta, float* __restrict__ bn_stat) {
  int t = threadIdx.x;
  if (t < 64) {
    float n   = (float)BN_ * 65.f;
    float mu  = bn_acc[t] / n;
    float var = bn_acc[64 + t] / n - mu * mu;
    float inv = rsqrtf(var + 1e-5f);
    float sc  = gamma[t] * inv;
    bn_stat[t]      = sc;
    bn_stat[64 + t] = beta[t] - mu * sc;
  }
}

// K3: normalized feats -> x_seq layout (S, B, FP=96) f16, zero-padded.
__global__ void k3_feats(const _Float16* __restrict__ HtH_h, const float* __restrict__ Hty,
                         const float* __restrict__ bn_stat, _Float16* __restrict__ feats) {
  int idx = blockIdx.x * 256 + threadIdx.x;   // (b, t)
  int b = idx >> 6, t = idx & 63;
  float sc = bn_stat[t], sh = bn_stat[64 + t];
  const _Float16* src = HtH_h + ((size_t)b * 64 + t) * 64;
  _Float16* dst = feats + ((size_t)t * BN_ + b) * FP_;
  for (int u = 0; u < 64; ++u) dst[u] = (_Float16)((float)src[u] * sc + sh);
  dst[64] = (_Float16)(Hty[(size_t)b * 64 + t] * sc + sh);
  for (int u = 65; u < FP_; ++u) dst[u] = (_Float16)0.f;
}

// ---------------------------------------------------------------------------
// K4: one GRU time step, both directions (grid.y). Block = 32 batch rows.
//     xg = x_t @ wih^T (K=96) and hg = h_{t-1} @ whh^T (K=128) via WMMA,
//     staged to LDS (2 x 32x384 f32), then gate nonlinearities; h_t stored
//     f16 into the per-direction output history (which doubles as h state).
// ---------------------------------------------------------------------------
__global__ void __launch_bounds__(256) k4_step(
    const _Float16* __restrict__ feats, const _Float16* __restrict__ wih_pack,
    const _Float16* __restrict__ whh_pack, const float* __restrict__ bih,
    const float* __restrict__ bhh, _Float16* __restrict__ go0,
    _Float16* __restrict__ go1, int s) {
  extern __shared__ float sm4[];
  float* xs = sm4;              // 32 x 384
  float* hs = sm4 + 32 * G3_;   // 32 x 384
  int d  = blockIdx.y;
  _Float16* go = d ? go1 : go0;
  int ss = d ? (SQ_ - 1 - s) : s;     // bwd direction reads reversed sequence
  int bb = blockIdx.x * 32;
  int sp = (s > 0) ? (s - 1) : 0;
  const _Float16* xbase = feats + ((size_t)ss * BN_ + bb) * FP_;
  const _Float16* hbase = go + ((size_t)sp * BN_ + bb) * HID_;
  const _Float16* wp = wih_pack + (size_t)d * 3 * 24 * 512;
  const _Float16* hp = whh_pack + (size_t)d * 4 * 24 * 512;
  __builtin_prefetch(wp, 0, 0);
  int wave = threadIdx.x >> 5;
#pragma unroll
  for (int j = 0; j < 6; ++j) {       // 48 tiles (2 x 24) over 8 waves
    int tile = wave * 6 + j;
    int mi = tile / 24, ni = tile % 24;
    v8f ax = {};
#pragma unroll
    for (int kt = 0; kt < 3; ++kt) {
      v16h a  = load_fragA_rm(xbase, mi * 16, FP_, kt);
      v16h bf = load_fragB_pack(wp, kt, ni, 24);
      ax = wmma16(a, bf, ax);
    }
    v8f ah = {};
    if (s > 0) {
#pragma unroll
      for (int kt = 0; kt < 4; ++kt) {
        v16h a  = load_fragA_rm(hbase, mi * 16, HID_, kt);
        v16h bf = load_fragB_pack(hp, kt, ni, 24);
        ah = wmma16(a, bf, ah);
      }
    }
    store_acc_f32(xs, mi * 16, ni * 16, G3_, ax);
    store_acc_f32(hs, mi * 16, ni * 16, G3_, ah);
  }
  __syncthreads();
  const float* bi = bih + d * G3_;
  const float* bh = bhh + d * G3_;
#pragma unroll
  for (int rep = 0; rep < 16; ++rep) {    // 32*128 h elements / 256 threads
    int e = threadIdx.x + rep * 256;
    int m = e >> 7, j = e & 127;
    float xr = xs[m * G3_ + j]       + bi[j];
    float hr = hs[m * G3_ + j]       + bh[j];
    float xz = xs[m * G3_ + 128 + j] + bi[128 + j];
    float hz = hs[m * G3_ + 128 + j] + bh[128 + j];
    float xn = xs[m * G3_ + 256 + j] + bi[256 + j];
    float hn = hs[m * G3_ + 256 + j] + bh[256 + j];
    float r = 1.f / (1.f + __expf(-(xr + hr)));
    float z = 1.f / (1.f + __expf(-(xz + hz)));
    float n = tanhf(xn + r * hn);
    float hold = (s > 0) ? (float)hbase[(size_t)m * HID_ + j] : 0.f;
    float hnew = (1.f - z) * n + z * hold;
    go[((size_t)s * BN_ + bb + m) * HID_ + j] = (_Float16)hnew;
  }
}

// ---------------------------------------------------------------------------
// K5: w_x[b][s][d] = go_orig[s,b,:] . Wx[d,:]  (K=256: fwd cols 0-127 from
//     go0[s], bwd cols 128-255 from go1[63-s]). One 16x16 tile per wave.
// ---------------------------------------------------------------------------
__global__ void __launch_bounds__(256) k5_wx(
    const _Float16* __restrict__ go0, const _Float16* __restrict__ go1,
    const _Float16* __restrict__ wx_pack, _Float16* __restrict__ wx_h) {
  int s  = blockIdx.y;
  int bb = blockIdx.x * 32;
  int wave = threadIdx.x >> 5;
  int mi = wave >> 2, ni = wave & 3;
  const _Float16* f0 = go0 + ((size_t)s * BN_ + bb) * HID_;
  const _Float16* f1 = go1 + ((size_t)(SQ_ - 1 - s) * BN_ + bb) * HID_;
  v8f acc = {};
#pragma unroll
  for (int kt = 0; kt < 8; ++kt) {
    const _Float16* src = (kt < 4) ? f0 : f1;
    v16h a  = load_fragA_rm(src, mi * 16, HID_, kt & 3);
    v16h bf = load_fragB_pack(wx_pack, kt, ni, 4);
    acc = wmma16(a, bf, acc);
  }
  int lane = threadIdx.x & 31;
  int dc   = ni * 16 + (lane & 15);
  int mo   = (lane < 16) ? 0 : 8;
#pragma unroll
  for (int i = 0; i < 8; ++i) {
    int m = mi * 16 + mo + i;
    wx_h[(((size_t)(bb + m)) * SQ_ + s) * 64 + dc] = (_Float16)acc[i];
  }
}

// K6: z0[b] = Wz @ (sum_s go_orig[s,b,:] * xhat0[b,s]).
__global__ void k6_z0(const _Float16* __restrict__ go0, const _Float16* __restrict__ go1,
                      const float* __restrict__ x_raw, const float* __restrict__ Wz,
                      float* __restrict__ zbuf) {
  __shared__ float xh[64];
  __shared__ float g[256];
  int b = blockIdx.x, tid = threadIdx.x;
  if (tid < 64) xh[tid] = 2.f * x_raw[(size_t)b * 64 + tid] - 3.f;  // 2^RATE-1 = 3
  __syncthreads();
  {
    float s = 0.f;
    if (tid < 128) {
      for (int t = 0; t < SQ_; ++t)
        s += (float)go0[((size_t)t * BN_ + b) * HID_ + tid] * xh[t];
    } else {
      int cc = tid - 128;
      for (int t = 0; t < SQ_; ++t)
        s += (float)go1[((size_t)(SQ_ - 1 - t) * BN_ + b) * HID_ + cc] * xh[t];
    }
    g[tid] = s;
  }
  __syncthreads();
  if (tid < 64) {
    float s = 0.f;
    const float* wr = Wz + (size_t)tid * 256;
    for (int c = 0; c < 256; ++c) s += g[c] * wr[c];
    zbuf[(size_t)b * 64 + tid] = s;
  }
}

// ---------------------------------------------------------------------------
// K7: per batch: T1 = HtH @ w_x ; whhw = w_x^T @ T1 (f32 out); wxt_hty.
//     All operands staged in WMMA-ready layouts (no strided gathers).
// ---------------------------------------------------------------------------
__global__ void __launch_bounds__(256) k7_whhw(
    const _Float16* __restrict__ HtH_h, const _Float16* __restrict__ wx_h,
    const float* __restrict__ Hty, float* __restrict__ whhw, float* __restrict__ wht) {
  __shared__ __align__(16) _Float16 WlT[64 * 64];  // w_x^T row-major (d x t): A layout
  __shared__ __align__(16) _Float16 Wpk[64 * 64];  // w_x in B-fragment pack
  __shared__ __align__(16) _Float16 Tpk[64 * 64];  // T1 in B-fragment pack
  __shared__ float htyl[64];
  int b = blockIdx.x, tid = threadIdx.x;
  const _Float16* wb = wx_h + (size_t)b * 4096;
  for (int i = tid; i < 4096; i += 256) {
    _Float16 v = wb[i];
    int t = i >> 6, d = i & 63;
    WlT[d * 64 + t] = v;
    int kt = t >> 5, ui = d >> 4;
    int pl = ((t >> 4) & 1) * 16 + (d & 15);
    Wpk[((kt * 4 + ui) * 32 + pl) * 16 + (t & 15)] = v;
  }
  if (tid < 64) htyl[tid] = Hty[(size_t)b * 64 + tid];
  __syncthreads();
  const _Float16* Ab = HtH_h + (size_t)b * 4096;
  int wave = tid >> 5;
#pragma unroll
  for (int rep = 0; rep < 2; ++rep) {            // T1 = HtH @ w_x
    int tile = wave * 2 + rep;
    int ti = tile >> 2, ui = tile & 3;
    v8f acc = {};
#pragma unroll
    for (int kt = 0; kt < 2; ++kt) {
      v16h a  = load_fragA_rm(Ab, ti * 16, 64, kt);   // HtH rows (global)
      v16h bf = load_fragB_pack(Wpk, kt, ui, 4);
      acc = wmma16(a, bf, acc);
    }
    store_acc_Bpack(Tpk, ti * 16, ui * 16, acc);      // straight into B layout
  }
  if (tid < 64) {                    // wxt_hty[d] = sum_t w_x[t][d]*Hty[t]
    float s = 0.f;
    for (int t = 0; t < 64; ++t) s += (float)WlT[tid * 64 + t] * htyl[t];
    wht[(size_t)b * 64 + tid] = s;
  }
  __syncthreads();
  float* ob = whhw + (size_t)b * 4096;
#pragma unroll
  for (int rep = 0; rep < 2; ++rep) {            // whhw = w_x^T @ T1
    int tile = wave * 2 + rep;
    int ti = tile >> 2, ui = tile & 3;
    v8f acc = {};
#pragma unroll
    for (int kt = 0; kt < 2; ++kt) {
      v16h a  = load_fragA_rm(WlT, ti * 16, 64, kt);
      v16h bf = load_fragB_pack(Tpk, kt, ui, 4);
      acc = wmma16(a, bf, acc);
    }
    store_acc_f32(ob, ti * 16, ui * 16, 64, acc);
  }
}

// K8: 10 GD iterations + final out = w_x @ z_hat. One 2-wave block per batch.
__global__ void __launch_bounds__(64) k8_gd(
    const float* __restrict__ whhw, const float* __restrict__ wht,
    const _Float16* __restrict__ wx_h, const float* __restrict__ zbuf,
    float* __restrict__ out) {
  __shared__ float W[64 * 64];
  __shared__ float zc[64];
  __shared__ float wh[64];
  int b = blockIdx.x, tid = threadIdx.x;
  const float* wb = whhw + (size_t)b * 4096;
  for (int i = tid; i < 4096; i += 64) W[i] = wb[i];
  zc[tid] = zbuf[(size_t)b * 64 + tid];
  wh[tid] = wht[(size_t)b * 64 + tid];
  __syncthreads();
  for (int it = 0; it < 10; ++it) {
    float nd = 0.f;
    const float* wr = W + tid * 64;
    for (int e = 0; e < 64; ++e) nd += wr[e] * zc[e];
    __syncthreads();
    zc[tid] = zc[tid] + 2e-6f * (wh[tid] - nd);   // 2*GD_STEP
    __syncthreads();
  }
  const _Float16* wxb = wx_h + ((size_t)b * 64 + tid) * 64;
  float s = 0.f;
  for (int d = 0; d < 64; ++d) s += (float)wxb[d] * zc[d];
  out[(size_t)b * 64 + tid] = s;
}

// ---------------------------------------------------------------------------
extern "C" void kernel_launch(void* const* d_in, const int* in_sizes, int n_in,
                              void* d_out, int out_size, void* d_ws, size_t ws_size,
                              hipStream_t stream) {
  const float* y     = (const float*)d_in[0];
  const float* H     = (const float*)d_in[1];
  const float* x_raw = (const float*)d_in[2];
  const float* wih   = (const float*)d_in[3];
  const float* whh   = (const float*)d_in[4];
  const float* bih   = (const float*)d_in[5];
  const float* bhh   = (const float*)d_in[6];
  const float* Wz    = (const float*)d_in[7];
  const float* Wx    = (const float*)d_in[8];
  const float* gamma = (const float*)d_in[9];
  const float* beta  = (const float*)d_in[10];
  float* out = (float*)d_out;

  char*  ws  = (char*)d_ws;
  size_t off = 0;
  auto take = [&](size_t bytes) -> char* {
    char* p = ws + off;
    off = (off + bytes + 255) & ~(size_t)255;
    return p;
  };
  _Float16* HtH_h    = (_Float16*)take((size_t)BN_ * 4096 * 2);
  float*    Hty      = (float*)   take((size_t)BN_ * 64 * 4);
  float*    bn_acc   = (float*)   take(512);
  float*    bn_stat  = (float*)   take(512);
  _Float16* feats    = (_Float16*)take((size_t)SQ_ * BN_ * FP_ * 2);
  _Float16* go0      = (_Float16*)take((size_t)SQ_ * BN_ * HID_ * 2);
  _Float16* go1      = (_Float16*)take((size_t)SQ_ * BN_ * HID_ * 2);
  _Float16* wih_pack = (_Float16*)take((size_t)NIH_ * 2);
  _Float16* whh_pack = (_Float16*)take((size_t)NHH_ * 2);
  _Float16* wx_pack  = (_Float16*)take((size_t)NWX_ * 2);
  _Float16* wx_h     = (_Float16*)take((size_t)BN_ * 4096 * 2);
  float*    whhw     = (float*)   take((size_t)BN_ * 4096 * 4);
  float*    wht      = (float*)   take((size_t)BN_ * 64 * 4);
  float*    zbuf     = (float*)   take((size_t)BN_ * 64 * 4);

  (void)hipFuncSetAttribute((const void*)k4_step,
                            hipFuncAttributeMaxDynamicSharedMemorySize, 96 * 1024);

  int k0n = 128 + NIH_ + NHH_ + NWX_;
  k0_pack<<<(k0n + 255) / 256, 256, 0, stream>>>(wih, whh, Wx, wih_pack, whh_pack,
                                                 wx_pack, bn_acc);
  k1_prep<<<BN_, 256, 0, stream>>>(H, y, HtH_h, Hty, bn_acc);
  k2_bn<<<1, 128, 0, stream>>>(bn_acc, gamma, beta, bn_stat);
  k3_feats<<<(BN_ * 64) / 256, 256, 0, stream>>>(HtH_h, Hty, bn_stat, feats);

  for (int s = 0; s < SQ_; ++s)
    k4_step<<<dim3(BN_ / 32, 2), 256, 96 * 1024, stream>>>(
        feats, wih_pack, whh_pack, bih, bhh, go0, go1, s);

  k5_wx<<<dim3(BN_ / 32, SQ_), 256, 0, stream>>>(go0, go1, wx_pack, wx_h);
  k6_z0<<<BN_, 256, 0, stream>>>(go0, go1, x_raw, Wz, zbuf);
  k7_whhw<<<BN_, 256, 0, stream>>>(HtH_h, wx_h, Hty, whhw, wht);
  k8_gd<<<BN_, 64, 0, stream>>>(whhw, wht, wx_h, zbuf, out);
}
